// FastScaledDotProductAttention_1168231104792
// MI455X (gfx1250) — compile-verified, matrix operand order checked by reading, not run on hardware
//
#include <hip/hip_runtime.h>
#include <hip/hip_bf16.h>
#include <stdint.h>

#define DIM 256
#define BATCH 32
#define MM 2048
#define NN 2048

typedef __attribute__((ext_vector_type(16))) _Float16 v16h;
typedef __attribute__((ext_vector_type(8)))  _Float16 v8h;
typedef __attribute__((ext_vector_type(8)))  float    v8f;
typedef __attribute__((ext_vector_type(4)))  unsigned int u32x4;
typedef __attribute__((ext_vector_type(4)))  int          i32x4;
typedef __attribute__((ext_vector_type(8)))  int          i32x8;

#define TDM_OK __has_builtin(__builtin_amdgcn_tensor_load_to_lds)

__device__ __forceinline__ v8f wmma_f16(v16h a, v16h b, v8f c) {
  // D = A(16x32 f16) * B(32x16 f16) + C(16x16 f32)
  return __builtin_amdgcn_wmma_f32_16x16x32_f16(false, a, false, b, (short)0, c,
                                                false, false);
}

__device__ __forceinline__ v8f zero8() {
  v8f z;
#pragma unroll
  for (int r = 0; r < 8; ++r) z[r] = 0.0f;
  return z;
}

__device__ __forceinline__ v8h cvt8(float4 a, float4 b) {
  v8h h;
  h[0] = (_Float16)a.x; h[1] = (_Float16)a.y; h[2] = (_Float16)a.z; h[3] = (_Float16)a.w;
  h[4] = (_Float16)b.x; h[5] = (_Float16)b.y; h[6] = (_Float16)b.z; h[7] = (_Float16)b.w;
  return h;
}

// Fragment from row-major f16 memory (row stride ld, K contiguous).
// Per ISA 16-bit A layout: row = lane%16, element j holds K = j + 8*((j>>3)+(lane>>4)).
// Same mapping serves B fragments when "row" is the B column and K is contiguous.
__device__ __forceinline__ v16h frag_kcontig(const _Float16* __restrict__ base,
                                             int ld, int lane) {
  const _Float16* p = base + (size_t)(lane & 15) * ld + ((lane >> 4) << 3);
  v8h lo = *(const v8h*)p;         // K = 0..7   (+8 for upper half-wave)
  v8h hi = *(const v8h*)(p + 16);  // K = 16..23 (+8)
  v16h f;
#pragma unroll
  for (int j = 0; j < 8; ++j) { f[j] = lo[j]; f[j + 8] = hi[j]; }
  return f;
}

// ---------------------------------------------------------------------------
// TDM: 2D tile (tile_d0 x tile_d1 elements of 2 bytes) global -> LDS with
// row stride stride0 (elements). LDS padding from descriptor: insert
// (pad_amt_code+1) DWORDs after every 2^(pad_int_code+1) DWORDs stored.
// ---------------------------------------------------------------------------
__device__ __forceinline__ void tdm_load_2d_f16(unsigned lds_off, const void* gaddr,
                                                unsigned tensor_d0, unsigned tensor_d1,
                                                unsigned tile_d0, unsigned tile_d1,
                                                unsigned stride0,
                                                unsigned pad_int_code,
                                                unsigned pad_amt_code) {
#if TDM_OK
  unsigned long long ga = (unsigned long long)(uintptr_t)gaddr;
  u32x4 g0;
  g0[0] = 1u;                                    // count=1 (valid user descriptor)
  g0[1] = lds_off;                               // lds_addr (bytes)
  g0[2] = (unsigned)(ga & 0xFFFFFFFFu);          // global_addr[31:0]
  g0[3] = (unsigned)((ga >> 32) & 0x01FFFFFFu) | (2u << 30);  // addr[56:32], type=2
  i32x8 g1;
  // wg_mask=0 | data_size=1(2B)<<16 | pad_enable<<20 | pad_interval<<22 | pad_amount<<25
  g1[0] = (int)((1u << 16) | (1u << 20) | (pad_int_code << 22) | (pad_amt_code << 25));
  g1[1] = (int)(tensor_d0 << 16);                         // tensor_dim0[15:0]
  g1[2] = (int)((tensor_d0 >> 16) | (tensor_d1 << 16));   // dim0 hi | dim1 lo
  g1[3] = (int)((tensor_d1 >> 16) | (tile_d0 << 16));     // dim1 hi | tile_dim0
  g1[4] = (int)(tile_d1 & 0xFFFFu);                       // tile_dim1 (tile_dim2=0)
  g1[5] = (int)stride0;                                   // tensor_dim0_stride lo
  g1[6] = 0;                                              // stride0 hi | stride1 lo
  g1[7] = 0;                                              // stride1 hi
  i32x4 z4 = {0, 0, 0, 0};
#if defined(__clang_major__) && __clang_major__ >= 23
  i32x8 z8 = {0, 0, 0, 0, 0, 0, 0, 0};
  __builtin_amdgcn_tensor_load_to_lds(g0, g1, z4, z4, z8, 0);
#else
  __builtin_amdgcn_tensor_load_to_lds(g0, g1, z4, z4, 0);
#endif
#endif
}

// ---------------------------------------------------------------------------
// Kernel 0: Wt[e][c] = (f16) W[c][e]
// ---------------------------------------------------------------------------
__global__ __launch_bounds__(256) void wt_kernel(const float* __restrict__ W,
                                                 _Float16* __restrict__ Wt) {
  int e = blockIdx.x, c = threadIdx.x;
  Wt[(size_t)e * DIM + c] = (_Float16)W[(size_t)c * DIM + e];
}

// ---------------------------------------------------------------------------
// Kernel 1: phi = relu(alpha * Wt @ X); software-pipelined f32->f16 staging.
// Q result stored transposed phiQT[b][m][e] (one packed b128 store per tile),
// K result stored phiK[b][e][n] (what kv/rowsum want).
// ---------------------------------------------------------------------------
__global__ __launch_bounds__(256) void phi_kernel(const float* __restrict__ query,
                                                  const float* __restrict__ key,
                                                  const _Float16* __restrict__ Wt,
                                                  _Float16* __restrict__ phiQT,
                                                  _Float16* __restrict__ phiK) {
  __shared__ _Float16 Xs[32][136];
  const int tid = threadIdx.x, lane = tid & 31, wid = tid >> 5;
  const int mblk = blockIdx.x;
  const int eblk = blockIdx.y;
  const int b    = blockIdx.z >> 1;
  const int isK  = blockIdx.z & 1;
  const float* __restrict__ X = isK ? key : query;
  const float alpha = isK ? 1.0f : (1.0f / 16.0f);

  const int e_wave  = eblk * 128 + (wid & 3) * 32;
  const int m_wave0 = (wid >> 2) * 64;
  const int m0      = mblk * 128;

  v8f acc[2][4];
#pragma unroll
  for (int i = 0; i < 2; ++i)
#pragma unroll
    for (int t = 0; t < 4; ++t) acc[i][t] = zero8();

  const int srow = tid >> 3;
  const int scol = (tid & 7) << 4;
  const float* src = X + (size_t)b * DIM * MM + m0 + scol;

  const float4* sp = (const float4*)(src + (size_t)srow * MM);
  float4 f0 = sp[0], f1 = sp[1], f2 = sp[2], f3 = sp[3];

  for (int c0 = 0; c0 < DIM; c0 += 32) {
    *(v8h*)&Xs[srow][scol]     = cvt8(f0, f1);
    *(v8h*)&Xs[srow][scol + 8] = cvt8(f2, f3);
    __syncthreads();

    if (c0 + 32 < DIM) {  // prefetch next k-step; latency overlaps WMMA below
      const float4* np = (const float4*)(src + (size_t)(c0 + 32 + srow) * MM);
      f0 = np[0]; f1 = np[1]; f2 = np[2]; f3 = np[3];
    }

    v16h a0 = frag_kcontig(Wt + (size_t)e_wave * DIM + c0, DIM, lane);
    v16h a1 = frag_kcontig(Wt + (size_t)(e_wave + 16) * DIM + c0, DIM, lane);
#pragma unroll
    for (int t = 0; t < 4; ++t) {
      v16h bf;
      int col = m_wave0 + t * 16 + (lane & 15);
      int kb  = (lane >> 4) << 3;
#pragma unroll
      for (int j = 0; j < 8; ++j) {
        bf[j]     = Xs[kb + j][col];
        bf[j + 8] = Xs[kb + 16 + j][col];
      }
      acc[0][t] = wmma_f16(a0, bf, acc[0][t]);
      acc[1][t] = wmma_f16(a1, bf, acc[1][t]);
    }
    __syncthreads();
  }

  if (!isK) {
    // phiQT[m][e]: lane's 8 acc elements are consecutive in e -> one b128 store
#pragma unroll
    for (int i = 0; i < 2; ++i)
#pragma unroll
      for (int t = 0; t < 4; ++t) {
        v8h h;
#pragma unroll
        for (int r = 0; r < 8; ++r) {
          float v = acc[i][t][r] * alpha;
          h[r] = (_Float16)(v > 0.0f ? v : 0.0f);
        }
        int e = e_wave + i * 16 + ((lane >> 4) << 3);
        int m = m0 + m_wave0 + t * 16 + (lane & 15);
        *(v8h*)(phiQT + ((size_t)b * MM + m) * DIM + e) = h;
      }
  } else {
#pragma unroll
    for (int i = 0; i < 2; ++i)
#pragma unroll
      for (int t = 0; t < 4; ++t)
#pragma unroll
        for (int r = 0; r < 8; ++r) {
          int e = e_wave + i * 16 + r + ((lane >> 4) << 3);
          int n = m0 + m_wave0 + t * 16 + (lane & 15);
          float v = acc[i][t][r];
          v = v > 0.0f ? v : 0.0f;
          phiK[((size_t)b * DIM + e) * NN + n] = (_Float16)v;
        }
  }
}

// ---------------------------------------------------------------------------
// Kernel 2: rowsum[b][e] = sum_n phiK[b][e][n]
// ---------------------------------------------------------------------------
__global__ __launch_bounds__(256) void rowsum_kernel(const _Float16* __restrict__ phiK,
                                                     float* __restrict__ rowsum) {
  __shared__ float red[256];
  int e = blockIdx.x, b = blockIdx.y, tid = threadIdx.x;
  const _Float16* row = phiK + ((size_t)b * DIM + e) * NN;
  float s = 0.0f;
  for (int i = tid; i < NN; i += 256) s += (float)row[i];
  red[tid] = s;
  __syncthreads();
  for (int off = 128; off > 0; off >>= 1) {
    if (tid < off) red[tid] += red[tid + off];
    __syncthreads();
  }
  if (tid == 0) rowsum[b * DIM + e] = red[0];
}

// ---------------------------------------------------------------------------
// Kernel 3: kvT[b][d][e] = sum_n phiK[b][e][n]*value[b][d][n]
// A tile (f16 phiK) staged by the Tensor Data Mover, double-buffered; the
// descriptor pads every 16 DW by 4 DW so rows land 40 halves apart (16B
// aligned, bank-conflict-free b128 fragment loads). B (value f32->f16) uses
// register-prefetch pipelining. TENSORcnt + barriers synchronize.
// ---------------------------------------------------------------------------
__global__ __launch_bounds__(256) void kv_kernel(const _Float16* __restrict__ phiK,
                                                 const float* __restrict__ value,
                                                 _Float16* __restrict__ kvT) {
  __shared__ _Float16 Vs[128][40];
#if TDM_OK
  __shared__ _Float16 As[2][128][40];
#endif
  const int tid = threadIdx.x, lane = tid & 31, wid = tid >> 5;
  const int dblk = blockIdx.x, eblk = blockIdx.y, b = blockIdx.z;
  const int e_wave  = eblk * 128 + (wid & 3) * 32;
  const int d_wave0 = (wid >> 2) * 64;
  const int d0      = dblk * 128;

  v8f acc[2][4];
#pragma unroll
  for (int i = 0; i < 2; ++i)
#pragma unroll
    for (int t = 0; t < 4; ++t) acc[i][t] = zero8();

  const int srow = tid >> 1;
  const int scol = (tid & 1) << 4;
  const float* src = value + ((size_t)b * DIM + d0 + srow) * NN + scol;

  const float4* sp = (const float4*)src;
  float4 f0 = sp[0], f1 = sp[1], f2 = sp[2], f3 = sp[3];

#if TDM_OK
  const _Float16* aGlob = phiK + ((size_t)b * DIM + eblk * 128) * NN;
  const unsigned aoff0 = (unsigned)(uintptr_t)(&As[0][0][0]);
  const unsigned aoff1 = (unsigned)(uintptr_t)(&As[1][0][0]);
  if (wid == 0)  // A tile for step 0: 32n x 128e rows, row stride NN
    tdm_load_2d_f16(aoff0, aGlob, NN, DIM, 32, 128, NN, /*int16DW*/3, /*amt4DW*/3);
#endif

  for (int n0 = 0; n0 < NN; n0 += 32) {
    *(v8h*)&Vs[srow][scol]     = cvt8(f0, f1);
    *(v8h*)&Vs[srow][scol + 8] = cvt8(f2, f3);
#if TDM_OK
    const int buf = (n0 >> 5) & 1;
    if (wid == 0) __builtin_amdgcn_s_wait_tensorcnt(0);
#endif
    __syncthreads();  // publish Vs stores (and DMA'd A tile)

    if (n0 + 32 < NN) {  // prefetch next B tile; overlaps WMMA below
      const float4* np = (const float4*)(src + n0 + 32);
      f0 = np[0]; f1 = np[1]; f2 = np[2]; f3 = np[3];
#if TDM_OK
      if (wid == 0)  // DMA next A tile into the other buffer
        tdm_load_2d_f16(buf ? aoff0 : aoff1, aGlob + n0 + 32, NN, DIM, 32, 128,
                        NN, 3, 3);
#endif
    }

#if TDM_OK
    v16h a0, a1;
    {
      int hw8 = (lane >> 4) << 3;
      const _Float16* p0 = &As[buf][(wid & 3) * 32 + (lane & 15)][hw8];
      const _Float16* p1 = &As[buf][(wid & 3) * 32 + 16 + (lane & 15)][hw8];
      v8h lo0 = *(const v8h*)p0, hi0 = *(const v8h*)(p0 + 16);
      v8h lo1 = *(const v8h*)p1, hi1 = *(const v8h*)(p1 + 16);
#pragma unroll
      for (int j = 0; j < 8; ++j) {
        a0[j] = lo0[j]; a0[j + 8] = hi0[j];
        a1[j] = lo1[j]; a1[j + 8] = hi1[j];
      }
    }
#else
    v16h a0 = frag_kcontig(phiK + ((size_t)b * DIM + e_wave) * NN + n0, NN, lane);
    v16h a1 = frag_kcontig(phiK + ((size_t)b * DIM + e_wave + 16) * NN + n0, NN, lane);
#endif
#pragma unroll
    for (int t = 0; t < 4; ++t) {
      const _Float16* rp = &Vs[d_wave0 + t * 16 + (lane & 15)][(lane >> 4) << 3];
      v8h lo = *(const v8h*)rp;
      v8h hi = *(const v8h*)(rp + 16);
      v16h bf;
#pragma unroll
      for (int j = 0; j < 8; ++j) { bf[j] = lo[j]; bf[j + 8] = hi[j]; }
      acc[0][t] = wmma_f16(a0, bf, acc[0][t]);
      acc[1][t] = wmma_f16(a1, bf, acc[1][t]);
    }
    __syncthreads();  // tiles consumed; buffers reusable
  }

#pragma unroll
  for (int i = 0; i < 2; ++i)
#pragma unroll
    for (int t = 0; t < 4; ++t)
#pragma unroll
      for (int r = 0; r < 8; ++r) {
        int e = e_wave + i * 16 + r + ((lane >> 4) << 3);
        int d = d0 + d_wave0 + t * 16 + (lane & 15);
        kvT[((size_t)b * DIM + d) * DIM + e] = (_Float16)acc[i][t][r];
      }
}

// ---------------------------------------------------------------------------
// Kernel 4: scale[b][m] = 1 / (sum_e phiQT[b][m][e]*rowsum[b][e] + 1e-8)
// phiQT is m-major: each thread reads 512 contiguous bytes (v8h loads)
// ---------------------------------------------------------------------------
__global__ __launch_bounds__(256) void scale_kernel(const _Float16* __restrict__ phiQT,
                                                    const float* __restrict__ rowsum,
                                                    float* __restrict__ scale) {
  int b = blockIdx.y;
  int m = blockIdx.x * 256 + threadIdx.x;
  const v8h* pq = (const v8h*)(phiQT + ((size_t)b * MM + m) * DIM);
  const float* rs = rowsum + b * DIM;
  float s = 0.0f;
#pragma unroll 4
  for (int c = 0; c < DIM / 8; ++c) {
    v8h h = pq[c];
#pragma unroll
    for (int j = 0; j < 8; ++j) s += (float)h[j] * rs[c * 8 + j];
  }
  scale[(size_t)b * MM + m] = 1.0f / (s + 1e-8f);
}

// ---------------------------------------------------------------------------
// Kernel 5: out[b][d][m] = scale[b][m] * sum_e kvT[b][d][e]*phiQT[b][m][e]
// Both operands K(=e)-contiguous f16 -> pure global b128 fragment loads +
// WMMA; no LDS, no barriers. kvT (4 MB) and phiQT tiles live in L2.
// ---------------------------------------------------------------------------
__global__ __launch_bounds__(256) void out_kernel(const _Float16* __restrict__ kvT,
                                                  const _Float16* __restrict__ phiQT,
                                                  const float* __restrict__ scale,
                                                  float* __restrict__ out) {
  const int tid = threadIdx.x, lane = tid & 31, wid = tid >> 5;
  const int mblk = blockIdx.x, dblk = blockIdx.y, b = blockIdx.z;
  const int d_wave  = dblk * 128 + (wid & 3) * 32;
  const int m_wave0 = (wid >> 2) * 64;
  const int m0      = mblk * 128;

  v8f acc[2][4];
#pragma unroll
  for (int i = 0; i < 2; ++i)
#pragma unroll
    for (int t = 0; t < 4; ++t) acc[i][t] = zero8();

  const _Float16* aBase = kvT + ((size_t)b * DIM + d_wave) * DIM;
  const _Float16* bBase = phiQT + ((size_t)b * MM + m0 + m_wave0) * DIM;

  for (int e0 = 0; e0 < DIM; e0 += 32) {
    v16h a0 = frag_kcontig(aBase + e0, DIM, lane);
    v16h a1 = frag_kcontig(aBase + (size_t)16 * DIM + e0, DIM, lane);
#pragma unroll
    for (int t = 0; t < 4; ++t) {
      v16h bf = frag_kcontig(bBase + (size_t)(t * 16) * DIM + e0, DIM, lane);
      acc[0][t] = wmma_f16(a0, bf, acc[0][t]);
      acc[1][t] = wmma_f16(a1, bf, acc[1][t]);
    }
  }

#pragma unroll
  for (int t = 0; t < 4; ++t) {
    int m = m0 + m_wave0 + t * 16 + (lane & 15);
    float sc = scale[(size_t)b * MM + m];
#pragma unroll
    for (int i = 0; i < 2; ++i)
#pragma unroll
      for (int r = 0; r < 8; ++r) {
        int d = d_wave + i * 16 + r + ((lane >> 4) << 3);
        out[((size_t)b * DIM + d) * MM + m] = acc[i][t][r] * sc;
      }
  }
}

// ---------------------------------------------------------------------------
extern "C" void kernel_launch(void* const* d_in, const int* in_sizes, int n_in,
                              void* d_out, int out_size, void* d_ws, size_t ws_size,
                              hipStream_t stream) {
  const float* query = (const float*)d_in[0];
  const float* key   = (const float*)d_in[1];
  const float* value = (const float*)d_in[2];
  const float* W     = (const float*)d_in[3];
  float* out = (float*)d_out;

  _Float16* Wt    = (_Float16*)d_ws;                       // 256*256 f16
  _Float16* phiQT = Wt + (size_t)DIM * DIM;                // [b][m][e], 32 MB
  _Float16* phiK  = phiQT + (size_t)BATCH * MM * DIM;      // [b][e][n], 32 MB
  _Float16* kvT   = phiK + (size_t)BATCH * DIM * NN;       // [b][d][e], 4 MB
  float* rowsum   = (float*)(kvT + (size_t)BATCH * DIM * DIM);
  float* scaleB   = rowsum + BATCH * DIM;

  wt_kernel<<<dim3(DIM), dim3(DIM), 0, stream>>>(W, Wt);
  phi_kernel<<<dim3(MM / 128, 2, BATCH * 2), dim3(256), 0, stream>>>(
      query, key, Wt, phiQT, phiK);
  rowsum_kernel<<<dim3(DIM, BATCH), dim3(256), 0, stream>>>(phiK, rowsum);
  kv_kernel<<<dim3(2, 2, BATCH), dim3(256), 0, stream>>>(phiK, value, kvT);
  scale_kernel<<<dim3(MM / 256, BATCH), dim3(256), 0, stream>>>(phiQT, rowsum, scaleB);
  out_kernel<<<dim3(MM / 128, 2, BATCH), dim3(256), 0, stream>>>(kvT, phiQT, scaleB, out);
}
